// DASE_51496657879113
// MI455X (gfx1250) — matrix operand order repaired and checked by reading, not compiled, not run on hardware
//
#include <hip/hip_runtime.h>
#include <math.h>

// ---------------------------------------------------------------- types
typedef __bf16 bf16;
typedef __attribute__((ext_vector_type(8)))  bf16  v8bf;
typedef __attribute__((ext_vector_type(16))) bf16  v16bf;
typedef __attribute__((ext_vector_type(8)))  float v8f;

#define B_N 8
#define C_N 256
#define H_N 64
#define W_N 64
#define HW_N (H_N*W_N)
#define PAD_N 2
#define P_N 25
#define K_N 819
#define HEADS_N 4
#define DH_N 64
#define DFF_N 1024
#define NTOK (B_N*K_N)        /* 6552   */
#define NPAT (NTOK*P_N)       /* 163800 */
#define CDIV(a,b) (((a)+(b)-1)/(b))

// -------------------------------------------- gfx1250 async-to-LDS path
#ifndef __has_builtin
#define __has_builtin(x) 0
#endif
#if defined(__HIP_DEVICE_COMPILE__) && __has_builtin(__builtin_amdgcn_global_load_async_to_lds_b128)
#define USE_ASYNC 1
#else
#define USE_ASYNC 0
#endif

#if USE_ASYNC
typedef int v4ia __attribute__((ext_vector_type(4)));   // builtin takes generic int4*
#if __has_builtin(__builtin_amdgcn_s_wait_asynccnt)
#define WAIT_ASYNC() __builtin_amdgcn_s_wait_asynccnt(0)
#else
#define WAIT_ASYNC() asm volatile("s_wait_asynccnt 0" ::: "memory")
#endif
#else
#define WAIT_ASYNC()
#endif

__device__ __forceinline__ float sigm(float v){ return 1.0f/(1.0f+expf(-v)); }
__device__ __forceinline__ int iclampd(int v,int lo,int hi){ return v<lo?lo:(v>hi?hi:v); }
__device__ __forceinline__ float decayf(int p){
  float gy = (float)(p/5 - 2), gx = (float)(p%5 - 2);
  return expf(-sqrtf(gy*gy+gx*gx)*(1.0f/2.5f));   // LAM*WIN == WIN*0.5 == 2.5
}
__device__ __forceinline__ v16bf ldfrag(const bf16* p){
  v16bf f;
  v8bf lo = *(const v8bf*)p;
  v8bf h2 = *(const v8bf*)(p + 8);
  for (int i = 0; i < 8; ++i) { f[i] = lo[i]; f[i+8] = h2[i]; }
  return f;
}

// ================================================================ GEMM
// out[M,N] = act( A[M,Kd] @ BT[N,Kd]^T + bias + resid ), bf16 in, f32 acc.
// A concat: k<ka0 from A0 (stride lda0), else A1 (row/a1div, stride lda1).
// Batched via gridDim.z with (z/hdiv, z%hdiv) strides.
// Per wave: 16x32 output tile (2 accumulators share the A fragment).

// ---- fast path: Kd%64==0 && ka0%64==0.  Branch-free inner loop,
// ---- K-step 64, double-buffered async LDS staging of the shared A tile.
__global__ void __launch_bounds__(256) gemm_fast(
    const bf16* A0, int ka0, int lda0, const bf16* A1, int a1div, int lda1,
    const bf16* BT, int ldb, const float* bias, const bf16* resid,
    float* outF, bf16* outB, int ldc,
    int M, int Kd, int N, int act,
    int hdiv, long sAb, long sAh, long sBb, long sBh, long sCb, long sCh)
{
  int z = blockIdx.z; int zb = z/hdiv, zh = z%hdiv;
  A0 += (long)zb*sAb + (long)zh*sAh;
  BT += (long)zb*sBb + (long)zh*sBh;
  long coff = (long)zb*sCb + (long)zh*sCh;

  __shared__ __attribute__((aligned(16))) bf16 Atile[2][16*64];

  int tid  = threadIdx.x;
  int wave = tid >> 5, lane = tid & 31;
  int l16 = lane & 15, hi = lane >> 4;
  int c0  = hi ? 16 : 0;                      // A: contiguous K halves
  int m0  = blockIdx.y * 16;
  int n0  = (blockIdx.x*8 + wave) * 32;
  int nr0 = n0 + l16;      if (nr0 >= N) nr0 = N-1;   // clamp, store masked
  int nr1 = n0 + 16 + l16; if (nr1 >= N) nr1 = N-1;
  const bf16* Bp0 = BT + (long)nr0*ldb;
  const bf16* Bp1 = BT + (long)nr1*ldb;

  v8f acc0, acc1;
  for (int i = 0; i < 8; ++i) { acc0[i] = 0.0f; acc1[i] = 0.0f; }

  auto stage = [&](int kbs, int buf) {        // 16 rows x 64 K, 128 x b128
    if (tid < 128) {
      int r = tid >> 3, q = tid & 7;
      int mr = m0 + r; if (mr >= M) mr = M-1;
      const bf16* abase; long rowi; int ld, kbr;
      if (kbs < ka0) { abase = A0; ld = lda0; rowi = mr;         kbr = kbs;       }
      else           { abase = A1; ld = lda1; rowi = mr / a1div; kbr = kbs - ka0; }
      const bf16* src = abase + rowi*(long)ld + kbr + q*8;
      bf16* dst = &Atile[buf][r*64 + q*8];
#if USE_ASYNC
      __builtin_amdgcn_global_load_async_to_lds_b128((v4ia*)src, (v4ia*)dst, 0, 0);
#else
      *(v8bf*)dst = *(const v8bf*)src;
#endif
    }
  };

  int cur = 0;
  stage(0, 0);
  for (int kb = 0; kb < Kd; kb += 64) {
    WAIT_ASYNC();
    __syncthreads();
    if (kb + 64 < Kd) stage(kb + 64, cur ^ 1);
    const bf16* At = &Atile[cur][0];
    #pragma unroll
    for (int ks = 0; ks < 64; ks += 32) {
      v16bf a;
      {
        v8bf lo = *(const v8bf*)&At[l16*64 + ks + c0];
        v8bf h2 = *(const v8bf*)&At[l16*64 + ks + c0 + 8];
        for (int i = 0; i < 8; ++i) { a[i] = lo[i]; a[i+8] = h2[i]; }
      }
      int kk = kb + ks + hi*16;               // B: contiguous K halves
      v16bf b0 = ldfrag(Bp0 + kk);
      v16bf b1 = ldfrag(Bp1 + kk);
      acc0 = __builtin_amdgcn_wmma_f32_16x16x32_bf16(false, a, false, b0, (short)0, acc0, false, false);
      acc1 = __builtin_amdgcn_wmma_f32_16x16x32_bf16(false, a, false, b1, (short)0, acc1, false, false);
    }
    cur ^= 1;
  }

  if (outF) outF += coff;
  if (outB) outB += coff;
  for (int half = 0; half < 2; ++half) {
    v8f accv = half ? acc1 : acc0;
    int nn = n0 + half*16 + l16;
    for (int r = 0; r < 8; ++r) {
      int m = m0 + r + 8*hi;
      if (m < M && nn < N) {
        float v = accv[r];
        if (bias)  v += bias[nn];
        if (resid) v += (float)resid[(long)m*ldc + nn];
        if (act == 1) v = fmaxf(v, 0.0f);
        else if (act == 2) v = sigm(v);
        long idx = (long)m*ldc + nn;
        if (outF) outF[idx] = v;
        if (outB) outB[idx] = (bf16)v;
      }
    }
  }
}

// ---- tail path: ragged Kd (att@v, Kd=819).  K-step 32, guarded tails.
__global__ void __launch_bounds__(256) gemm_tail(
    const bf16* A0, int ka0, int lda0, const bf16* A1, int a1div, int lda1,
    const bf16* BT, int ldb, const float* bias, const bf16* resid,
    float* outF, bf16* outB, int ldc,
    int M, int Kd, int N, int act,
    int hdiv, long sAb, long sAh, long sBb, long sBh, long sCb, long sCh)
{
  int z = blockIdx.z; int zb = z/hdiv, zh = z%hdiv;
  A0 += (long)zb*sAb + (long)zh*sAh;
  BT += (long)zb*sBb + (long)zh*sBh;
  long coff = (long)zb*sCb + (long)zh*sCh;

  __shared__ __attribute__((aligned(16))) bf16 Atile[16*32];

  int tid  = threadIdx.x;
  int wave = tid >> 5, lane = tid & 31;
  int l16 = lane & 15, hi = lane >> 4;
  int c0  = hi ? 16 : 0;
  int m0  = blockIdx.y * 16;
  int n0  = (blockIdx.x*8 + wave) * 32;
  int nr0 = n0 + l16;      if (nr0 >= N) nr0 = N-1;
  int nr1 = n0 + 16 + l16; if (nr1 >= N) nr1 = N-1;
  const bf16* Bp0 = BT + (long)nr0*ldb;
  const bf16* Bp1 = BT + (long)nr1*ldb;

  v8f acc0, acc1;
  for (int i = 0; i < 8; ++i) { acc0[i] = 0.0f; acc1[i] = 0.0f; }

  for (int kb = 0; kb < Kd; kb += 32) {
    __syncthreads();
    if (tid < 64) {
      int r = tid >> 2, q = tid & 3;
      int mr = m0 + r; if (mr >= M) mr = M-1;
      const bf16* abase; long rowi; int ld, kbr, Klen;
      if (kb < ka0) { abase = A0; ld = lda0; rowi = mr;         kbr = kb;       Klen = ka0;      }
      else          { abase = A1; ld = lda1; rowi = mr / a1div; kbr = kb - ka0; Klen = Kd - ka0; }
      int kr = kbr + q*8;
      bf16* dst = &Atile[r*32 + q*8];
      if (kr + 8 <= Klen) {
        const bf16* src = abase + rowi*(long)ld + kr;
#if USE_ASYNC
        __builtin_amdgcn_global_load_async_to_lds_b128((v4ia*)src, (v4ia*)dst, 0, 0);
#else
        *(v8bf*)dst = *(const v8bf*)src;
#endif
      } else {
        for (int j = 0; j < 8; ++j)
          dst[j] = (kr + j < Klen) ? abase[rowi*(long)ld + kr + j] : (bf16)0.0f;
      }
    }
    WAIT_ASYNC();
    __syncthreads();

    v16bf a;
    {
      v8bf lo = *(const v8bf*)&Atile[l16*32 + c0];
      v8bf h2 = *(const v8bf*)&Atile[l16*32 + c0 + 8];
      for (int i = 0; i < 8; ++i) { a[i] = lo[i]; a[i+8] = h2[i]; }
    }
    int kk = kb + hi*16;
    v16bf b0, b1;
    if (kb + 32 <= Kd) {                      // uniform branch
      b0 = ldfrag(Bp0 + kk);
      b1 = ldfrag(Bp1 + kk);
    } else {
      for (int i = 0; i < 16; ++i) {
        int k = kk + i;
        b0[i] = (k < Kd) ? Bp0[k] : (bf16)0.0f;
        b1[i] = (k < Kd) ? Bp1[k] : (bf16)0.0f;
      }
    }
    acc0 = __builtin_amdgcn_wmma_f32_16x16x32_bf16(false, a, false, b0, (short)0, acc0, false, false);
    acc1 = __builtin_amdgcn_wmma_f32_16x16x32_bf16(false, a, false, b1, (short)0, acc1, false, false);
  }

  if (outF) outF += coff;
  if (outB) outB += coff;
  for (int half = 0; half < 2; ++half) {
    v8f accv = half ? acc1 : acc0;
    int nn = n0 + half*16 + l16;
    for (int r = 0; r < 8; ++r) {
      int m = m0 + r + 8*hi;
      if (m < M && nn < N) {
        float v = accv[r];
        if (bias)  v += bias[nn];
        if (resid) v += (float)resid[(long)m*ldc + nn];
        if (act == 1) v = fmaxf(v, 0.0f);
        else if (act == 2) v = sigm(v);
        long idx = (long)m*ldc + nn;
        if (outF) outF[idx] = v;
        if (outB) outB[idx] = (bf16)v;
      }
    }
  }
}

// ====================================================== helper kernels
__global__ void zero_kernel(float* p, long n){
  long i = (long)blockIdx.x*256 + threadIdx.x; if (i < n) p[i] = 0.0f;
}
__global__ void f2b_kernel(const float* s, bf16* d, long n){
  long i = (long)blockIdx.x*256 + threadIdx.x; if (i < n) d[i] = (bf16)s[i];
}
// BT[n*Kin+k] = W[k*Nout+n]  (weight transpose + bf16 convert)
__global__ void transpose_w_kernel(const float* Wm, bf16* BT, int Kin, int Nout){
  long i = (long)blockIdx.x*256 + threadIdx.x;
  if (i >= (long)Kin*Nout) return;
  int n = (int)(i / Kin), k = (int)(i % Kin);
  BT[i] = (bf16)Wm[(long)k*Nout + n];
}
// combine 3x3/5x5/7x7 saliency weights into one 7x7, /3
__global__ void wcomb_kernel(const float* w3, const float* w5, const float* w7, float* wc){
  int i = blockIdx.x*256 + threadIdx.x;
  if (i >= C_N*49) return;
  int c = i / 49, t = i % 49, ky = t/7, kx = t%7;
  float w = w7[i];
  if (ky>=1 && ky<=5 && kx>=1 && kx<=5) w += w5[c*25 + (ky-1)*5 + (kx-1)];
  if (ky>=2 && ky<=4 && kx>=2 && kx<=4) w += w3[c*9  + (ky-2)*3 + (kx-2)];
  wc[i] = w*(1.0f/3.0f);
}
__global__ void saliency_kernel(const float* x, const float* wc,
                                const float* b3, const float* b5, const float* b7,
                                const float* maskl, float* logits){
  long i = (long)blockIdx.x*256 + threadIdx.x;
  if (i >= (long)B_N*HW_N) return;
  int b = (int)(i / HW_N), pix = (int)(i % HW_N);
  int y = pix / W_N, xx = pix % W_N;
  float acc = 0.0f;
  for (int c = 0; c < C_N; ++c) {
    const float* xc = x + ((long)b*C_N + c)*HW_N;
    const float* w  = wc + c*49;
    for (int t = 0; t < 49; ++t) {
      int yy = y + t/7 - 3, xw = xx + t%7 - 3;
      if (yy>=0 && yy<H_N && xw>=0 && xw<W_N) acc += xc[yy*W_N + xw]*w[t];
    }
  }
  logits[i] = acc + (b3[0]+b5[0]+b7[0])*(1.0f/3.0f) + maskl[pix];
}
__global__ void softmax_hw_kernel(const float* logits, float* soft){
  int b = blockIdx.x, c = threadIdx.x;
  const float* L = logits + (long)b*HW_N;
  __shared__ float red[256]; __shared__ float sM, sS;
  float mx = -3.0e38f;
  for (int i = c; i < HW_N; i += 256) mx = fmaxf(mx, L[i]*2.0f);     // /TEMP
  red[c] = mx; __syncthreads();
  for (int s = 128; s > 0; s >>= 1){ if (c < s) red[c] = fmaxf(red[c], red[c+s]); __syncthreads(); }
  if (c == 0) sM = red[0]; __syncthreads();
  float sum = 0.0f;
  for (int i = c; i < HW_N; i += 256) sum += expf(L[i]*2.0f - sM);
  red[c] = sum; __syncthreads();
  for (int s = 128; s > 0; s >>= 1){ if (c < s) red[c] += red[c+s]; __syncthreads(); }
  if (c == 0) sS = red[0]; __syncthreads();
  float inv = 1.0f/sS;
  float* O = soft + (long)b*HW_N;
  for (int i = c; i < HW_N; i += 256) O[i] = expf(L[i]*2.0f - sM)*inv;
}
// per-batch radix select of the K largest softmax probs (positive floats)
__global__ void topk_kernel(const float* soft, int* topk){
  int b = blockIdx.x;
  const float* s = soft + (long)b*HW_N;
  __shared__ unsigned hist[256];
  __shared__ unsigned sh_prefix; __shared__ int sh_rem;
  if (threadIdx.x == 0){ sh_prefix = 0u; sh_rem = K_N; }
  __syncthreads();
  for (int round = 3; round >= 0; --round) {
    hist[threadIdx.x] = 0u; __syncthreads();
    unsigned pfx = sh_prefix;
    int shift = round*8;
    unsigned mhi = (round == 3) ? 0u : (0xFFFFFFFFu << (8*(round+1)));
    for (int i = threadIdx.x; i < HW_N; i += 256) {
      unsigned bits = __float_as_uint(s[i]);
      if ((bits & mhi) == (pfx & mhi)) atomicAdd(&hist[(bits >> shift) & 255u], 1u);
    }
    __syncthreads();
    if (threadIdx.x == 0) {
      int acc = 0, rem = sh_rem, byte = 0;
      for (int v = 255; v >= 0; --v) {
        int h = (int)hist[v];
        if (acc + h >= rem) { byte = v; rem = rem - acc; break; }
        acc += h;
      }
      sh_rem = rem; sh_prefix = pfx | ((unsigned)byte << shift);
    }
    __syncthreads();
  }
  __shared__ int eq_cnt, sel_cnt;
  if (threadIdx.x == 0){ eq_cnt = 0; sel_cnt = 0; }
  __syncthreads();
  unsigned thr = sh_prefix; int rem = sh_rem;
  for (int i = threadIdx.x; i < HW_N; i += 256) {
    unsigned bits = __float_as_uint(s[i]);
    bool take = false;
    if (bits > thr) take = true;
    else if (bits == thr && atomicAdd(&eq_cnt, 1) < rem) take = true;
    if (take) { int p = atomicAdd(&sel_cnt, 1); if (p < K_N) topk[(long)b*K_N + p] = i; }
  }
}
__global__ void coords_kernel(const int* topk, int* posidx, int* lin){
  int r = blockIdx.x*256 + threadIdx.x;
  if (r >= NTOK) return;
  int id = topk[r];
  int ys = iclampd(id / W_N, PAD_N, H_N-1-PAD_N);
  int xs = iclampd(id % W_N, PAD_N, W_N-1-PAD_N);
  posidx[r] = ys*W_N + xs;
  for (int p = 0; p < P_N; ++p) {
    int yy = iclampd(ys + p/5 - 2, 0, H_N-1);
    int xx = iclampd(xs + p%5 - 2, 0, W_N-1);
    lin[(long)r*P_N + p] = yy*W_N + xx;
  }
}
__global__ void patch_build_kernel(const float* x, const float* soft, const int* lin,
                                   const float* gamma, bf16* patches){
  int r = blockIdx.x;                  // (b,k) token
  int b = r / K_N;
  __shared__ float sc[P_N]; __shared__ float smean;
  if (threadIdx.x < P_N)
    sc[threadIdx.x] = soft[(long)b*HW_N + lin[(long)r*P_N + threadIdx.x]];
  __syncthreads();
  if (threadIdx.x == 0){ float s = 0; for (int p = 0; p < P_N; ++p) s += sc[p]; smean = s*(1.0f/P_N); }
  __syncthreads();
  float gm = gamma[0];
  for (int idx = threadIdx.x; idx < P_N*C_N; idx += 256) {
    int p = idx / C_N, c = idx % C_N;
    int pix = lin[(long)r*P_N + p];
    float xv = x[((long)b*C_N + c)*HW_N + pix];
    float wv = sigm(gm*(sc[p] - smean)) * decayf(p);
    patches[((long)r*P_N + p)*C_N + c] = (bf16)(xv * wv);
  }
}
__global__ void pool_kernel(const bf16* loc, bf16* meanb, bf16* maxb, float* sump){
  long r = blockIdx.x; int c = threadIdx.x;
  float s = 0.0f, m = -3.0e38f;
  for (int p = 0; p < P_N; ++p) {
    float v = (float)loc[(r*P_N + p)*C_N + c];
    s += v; m = fmaxf(m, v);
  }
  sump[r*C_N + c]  = s;
  meanb[r*C_N + c] = (bf16)(s*(1.0f/P_N));
  maxb[r*C_N + c]  = (bf16)m;
}
__global__ void weighted_kernel(const float* aw, const float* sump, bf16* outb, long n){
  long i = (long)blockIdx.x*256 + threadIdx.x;
  if (i < n) outb[i] = (bf16)(aw[i]*sump[i]);       // aw * sum_p(local)
}
__global__ void posadd_kernel(float* token, const float* pos, const int* posidx, bf16* tbf, long n){
  long i = (long)blockIdx.x*256 + threadIdx.x;
  if (i >= n) return;
  long r = i / C_N; int c = (int)(i % C_N);
  float t = token[i] + pos[(long)posidx[r]*C_N + c];
  token[i] = t; tbf[i] = (bf16)t;
}
__global__ void vT_kernel(const bf16* v, bf16* vT){
  long i = (long)blockIdx.x*256 + threadIdx.x;
  if (i >= (long)B_N*HEADS_N*DH_N*K_N) return;
  int k = (int)(i % K_N); long t = i / K_N;
  int d = (int)(t % DH_N); t /= DH_N;
  int h = (int)(t % HEADS_N); int b = (int)(t / HEADS_N);
  vT[i] = v[((long)(b*K_N + k))*C_N + h*DH_N + d];
}
__global__ void attn_softmax_kernel(const float* scores, bf16* att){
  long row = blockIdx.x;               // (b,h,q)
  long base = row*K_N; int c = threadIdx.x;
  __shared__ float red[256]; __shared__ float sM, sS;
  float mx = -3.0e38f;
  for (int i = c; i < K_N; i += 256) mx = fmaxf(mx, scores[base+i]*0.125f);
  red[c] = mx; __syncthreads();
  for (int s = 128; s > 0; s >>= 1){ if (c < s) red[c] = fmaxf(red[c], red[c+s]); __syncthreads(); }
  if (c == 0) sM = red[0]; __syncthreads();
  float sum = 0.0f;
  for (int i = c; i < K_N; i += 256) sum += expf(scores[base+i]*0.125f - sM);
  red[c] = sum; __syncthreads();
  for (int s = 128; s > 0; s >>= 1){ if (c < s) red[c] += red[c+s]; __syncthreads(); }
  if (c == 0) sS = red[0]; __syncthreads();
  float inv = 1.0f/sS;
  for (int i = c; i < K_N; i += 256)
    att[base+i] = (bf16)(expf(scores[base+i]*0.125f - sM)*inv);
}
__global__ void ln_kernel(const float* a, const float* badd, const float* g, const float* be,
                          float* outF, bf16* outB){
  long row = blockIdx.x; int c = threadIdx.x;
  long idx = row*C_N + c;
  float val = a[idx] + (badd ? badd[idx] : 0.0f);
  __shared__ float red[256]; __shared__ float mu, rs;
  red[c] = val; __syncthreads();
  for (int s = 128; s > 0; s >>= 1){ if (c < s) red[c] += red[c+s]; __syncthreads(); }
  if (c == 0) mu = red[0]*(1.0f/C_N); __syncthreads();
  float d = val - mu;
  red[c] = d*d; __syncthreads();
  for (int s = 128; s > 0; s >>= 1){ if (c < s) red[c] += red[c+s]; __syncthreads(); }
  if (c == 0) rs = rsqrtf(red[0]*(1.0f/C_N) + 1e-5f); __syncthreads();
  float o = d*rs*g[c] + be[c];
  if (outF) outF[idx] = o;
  if (outB) outB[idx] = (bf16)o;
}
__global__ void fused_kernel(bf16* gbuf, const bf16* t2b, const bf16* loc, long n){
  long i = (long)blockIdx.x*256 + threadIdx.x;
  if (i >= n) return;
  long r = i / C_N; int c = (int)(i % C_N);
  long tok = r / P_N;
  float g = (float)gbuf[i];
  float f = g*(float)t2b[tok*C_N + c] + (1.0f - g)*(float)loc[i];
  gbuf[i] = (bf16)f;                   // in place: gbuf becomes "fused"
}
__global__ void pi2_kernel(const bf16* pih, const float* w2, const float* b2, float* imp){
  long r = blockIdx.x; int c = threadIdx.x;
  __shared__ float red[256];
  red[c] = (float)pih[r*C_N + c]*w2[c]; __syncthreads();
  for (int s = 128; s > 0; s >>= 1){ if (c < s) red[c] += red[c+s]; __syncthreads(); }
  if (c == 0)
    imp[r] = sigm(red[0] + b2[0]) * decayf((int)(r % P_N));
}
__global__ void scatter_kernel(const bf16* fused, const float* imp, const int* lin,
                               float* fs, float* wsum){
  long r = blockIdx.x; int c = threadIdx.x;
  int b = (int)(r / ((long)K_N*P_N));
  int pix = lin[r];
  float im = imp[r];
  long base = ((long)b*HW_N + pix)*C_N;
  atomicAdd(&fs[base + c], (float)fused[r*C_N + c]*im);
  if (c == 0) atomicAdd(&wsum[(long)b*HW_N + pix], im);
}
__global__ void final_kernel(const float* fs, const float* wsum, const float* x,
                             const float* g, const float* be, float* out){
  int row = blockIdx.x;                // b*HW + pix
  int b = row / HW_N, pix = row % HW_N;
  int c = threadIdx.x;
  float wv = fmaxf(wsum[row], 1e-6f);
  float val = fs[(long)row*C_N + c]/wv + x[((long)b*C_N + c)*HW_N + pix];
  __shared__ float red[256]; __shared__ float mu, rs;
  red[c] = val; __syncthreads();
  for (int s = 128; s > 0; s >>= 1){ if (c < s) red[c] += red[c+s]; __syncthreads(); }
  if (c == 0) mu = red[0]*(1.0f/C_N); __syncthreads();
  float d = val - mu;
  red[c] = d*d; __syncthreads();
  for (int s = 128; s > 0; s >>= 1){ if (c < s) red[c] += red[c+s]; __syncthreads(); }
  if (c == 0) rs = rsqrtf(red[0]*(1.0f/C_N) + 1e-5f); __syncthreads();
  out[(long)row*C_N + c] = d*rs*g[c] + be[c];
}

// ================================================================ host
static void launch_gemm(hipStream_t st,
    const bf16* A0, int ka0, int lda0, const bf16* A1, int a1div, int lda1,
    const bf16* BT, int ldb, const float* bias, const bf16* resid,
    float* outF, bf16* outB, int ldc, int M, int Kd, int N, int act,
    int batch = 1, int hdiv = 1, long sAb = 0, long sAh = 0,
    long sBb = 0, long sBh = 0, long sCb = 0, long sCh = 0)
{
  dim3 grid((unsigned)CDIV(N, 256), (unsigned)CDIV(M, 16), (unsigned)batch);
  if ((Kd % 64 == 0) && (ka0 % 64 == 0))
    gemm_fast<<<grid, 256, 0, st>>>(A0, ka0, lda0, A1, a1div, lda1, BT, ldb,
        bias, resid, outF, outB, ldc, M, Kd, N, act,
        hdiv, sAb, sAh, sBb, sBh, sCb, sCh);
  else
    gemm_tail<<<grid, 256, 0, st>>>(A0, ka0, lda0, A1, a1div, lda1, BT, ldb,
        bias, resid, outF, outB, ldc, M, Kd, N, act,
        hdiv, sAb, sAh, sBb, sBh, sCb, sCh);
}

extern "C" void kernel_launch(void* const* d_in, const int* in_sizes, int n_in,
                              void* d_out, int out_size, void* d_ws, size_t ws_size,
                              hipStream_t stream) {
  (void)in_sizes; (void)n_in; (void)out_size; (void)ws_size;

  const float* x     = (const float*)d_in[0];
  const float* sw3   = (const float*)d_in[1];  const float* sb3  = (const float*)d_in[2];
  const float* sw5   = (const float*)d_in[3];  const float* sb5  = (const float*)d_in[4];
  const float* sw7   = (const float*)d_in[5];  const float* sb7  = (const float*)d_in[6];
  const float* maskl = (const float*)d_in[7];  const float* gamma= (const float*)d_in[8];
  const float* lr_w1 = (const float*)d_in[9];  const float* lr_b1= (const float*)d_in[10];
  const float* lr_w2 = (const float*)d_in[11]; const float* lr_b2= (const float*)d_in[12];
  const float* ca_w1 = (const float*)d_in[13]; const float* ca_b1= (const float*)d_in[14];
  const float* ca_w2 = (const float*)d_in[15]; const float* ca_b2= (const float*)d_in[16];
  const float* fus_w = (const float*)d_in[17]; const float* fus_b= (const float*)d_in[18];
  const float* pos   = (const float*)d_in[19];
  const float* wq    = (const float*)d_in[20]; const float* bq   = (const float*)d_in[21];
  const float* wk    = (const float*)d_in[22]; const float* bk   = (const float*)d_in[23];
  const float* wv    = (const float*)d_in[24]; const float* bv   = (const float*)d_in[25];
  const float* wo    = (const float*)d_in[26]; const float* bo   = (const float*)d_in[27];
  const float* ln1_g = (const float*)d_in[28]; const float* ln1_b= (const float*)d_in[29];
  const float* ff_w1 = (const float*)d_in[30]; const float* ff_b1= (const float*)d_in[31];
  const float* ff_w2 = (const float*)d_in[32]; const float* ff_b2= (const float*)d_in[33];
  const float* ln2_g = (const float*)d_in[34]; const float* ln2_b= (const float*)d_in[35];
  const float* g_w   = (const float*)d_in[36]; const float* g_b  = (const float*)d_in[37];
  const float* pi_w1 = (const float*)d_in[38]; const float* pi_b1= (const float*)d_in[39];
  const float* pi_w2 = (const float*)d_in[40]; const float* pi_b2= (const float*)d_in[41];
  const float* norm_g= (const float*)d_in[42]; const float* norm_b=(const float*)d_in[43];

  // -------- workspace bump allocator --------
  char* wsbase = (char*)d_ws;
  size_t off = 0;
  auto alloc = [&](size_t bytes) -> char* {
    char* p = wsbase + off;
    off += (bytes + 255) & ~(size_t)255;
    return p;
  };
  float* wcomb  = (float*)alloc(sizeof(float)*C_N*49);
  float* logits = (float*)alloc(sizeof(float)*(size_t)B_N*HW_N);
  float* soft   = (float*)alloc(sizeof(float)*(size_t)B_N*HW_N);
  int*   topk   = (int*)  alloc(sizeof(int)*NTOK);
  int*   posidx = (int*)  alloc(sizeof(int)*NTOK);
  int*   lin    = (int*)  alloc(sizeof(int)*(size_t)NPAT);

  bf16* lr1T = (bf16*)alloc(2*(size_t)C_N*C_N);
  bf16* lr2T = (bf16*)alloc(2*(size_t)C_N*C_N);
  bf16* ca1T = (bf16*)alloc(2*(size_t)64*C_N);
  bf16* ca2T = (bf16*)alloc(2*(size_t)C_N*64);
  bf16* fusT = (bf16*)alloc(2*(size_t)C_N*512);
  bf16* wqT  = (bf16*)alloc(2*(size_t)C_N*C_N);
  bf16* wkT  = (bf16*)alloc(2*(size_t)C_N*C_N);
  bf16* wvT  = (bf16*)alloc(2*(size_t)C_N*C_N);
  bf16* woT  = (bf16*)alloc(2*(size_t)C_N*C_N);
  bf16* ff1T = (bf16*)alloc(2*(size_t)DFF_N*C_N);
  bf16* ff2T = (bf16*)alloc(2*(size_t)C_N*DFF_N);
  bf16* gT   = (bf16*)alloc(2*(size_t)C_N*512);
  bf16* pi1T = (bf16*)alloc(2*(size_t)C_N*C_N);

  size_t patE = (size_t)NPAT*C_N, tokE = (size_t)NTOK*C_N;
  bf16* patches = (bf16*)alloc(2*patE);
  bf16* hidden  = (bf16*)alloc(2*patE);
  bf16* localb  = (bf16*)alloc(2*patE);
  bf16* gbuf    = (bf16*)alloc(2*patE);

  bf16* meanb = (bf16*)alloc(2*tokE);
  bf16* maxb  = (bf16*)alloc(2*tokE);
  bf16* wgtb  = (bf16*)alloc(2*tokE);
  float* sump = (float*)alloc(4*tokE);
  bf16* cah   = (bf16*)alloc(2*(size_t)NTOK*64);
  float* aw   = (float*)alloc(4*tokE);
  float* token= (float*)alloc(4*tokE);
  bf16* tbf   = (bf16*)alloc(2*tokE);
  bf16* qb    = (bf16*)alloc(2*tokE);
  bf16* kbuf  = (bf16*)alloc(2*tokE);
  bf16* vb    = (bf16*)alloc(2*tokE);
  bf16* vT    = (bf16*)alloc(2*tokE);
  size_t attE = (size_t)B_N*HEADS_N*K_N*K_N;
  float* scores = (float*)alloc(4*attE);
  bf16*  att    = (bf16*)alloc(2*attE);
  float* obuf = (float*)alloc(4*tokE);
  bf16*  obf  = (bf16*)alloc(2*tokE);
  float* oproj= (float*)alloc(4*tokE);
  float* t1   = (float*)alloc(4*tokE);
  bf16*  t1b  = (bf16*)alloc(2*tokE);
  bf16*  ffh  = (bf16*)alloc(2*(size_t)NTOK*DFF_N);
  float* ff2o = (float*)alloc(4*tokE);
  float* t2   = (float*)alloc(4*tokE);
  bf16*  t2b  = (bf16*)alloc(2*tokE);
  float* imp  = (float*)alloc(4*(size_t)NPAT);
  float* fs   = (float*)alloc(4*(size_t)B_N*HW_N*C_N);
  float* wsum = (float*)alloc(4*(size_t)B_N*HW_N);

  // -------- stage 1: saliency -> softmax -> top-K -> coords --------
  wcomb_kernel<<<CDIV(C_N*49,256),256,0,stream>>>(sw3, sw5, sw7, wcomb);
  saliency_kernel<<<CDIV(B_N*HW_N,256),256,0,stream>>>(x, wcomb, sb3, sb5, sb7, maskl, logits);
  softmax_hw_kernel<<<B_N,256,0,stream>>>(logits, soft);
  topk_kernel<<<B_N,256,0,stream>>>(soft, topk);
  coords_kernel<<<CDIV(NTOK,256),256,0,stream>>>(topk, posidx, lin);

  // -------- weight transposes (f32 -> bf16, [N][K] layout) --------
  transpose_w_kernel<<<CDIV(C_N*C_N,256),256,0,stream>>>(lr_w1, lr1T, C_N, C_N);
  transpose_w_kernel<<<CDIV(C_N*C_N,256),256,0,stream>>>(lr_w2, lr2T, C_N, C_N);
  transpose_w_kernel<<<CDIV(C_N*64,256),256,0,stream>>>(ca_w1, ca1T, C_N, 64);
  transpose_w_kernel<<<CDIV(64*C_N,256),256,0,stream>>>(ca_w2, ca2T, 64, C_N);
  transpose_w_kernel<<<CDIV(512*C_N,256),256,0,stream>>>(fus_w, fusT, 512, C_N);
  transpose_w_kernel<<<CDIV(C_N*C_N,256),256,0,stream>>>(wq, wqT, C_N, C_N);
  transpose_w_kernel<<<CDIV(C_N*C_N,256),256,0,stream>>>(wk, wkT, C_N, C_N);
  transpose_w_kernel<<<CDIV(C_N*C_N,256),256,0,stream>>>(wv, wvT, C_N, C_N);
  transpose_w_kernel<<<CDIV(C_N*C_N,256),256,0,stream>>>(wo, woT, C_N, C_N);
  transpose_w_kernel<<<CDIV(C_N*DFF_N,256),256,0,stream>>>(ff_w1, ff1T, C_N, DFF_N);
  transpose_w_kernel<<<CDIV(DFF_N*C_N,256),256,0,stream>>>(ff_w2, ff2T, DFF_N, C_N);
  transpose_w_kernel<<<CDIV(512*C_N,256),256,0,stream>>>(g_w, gT, 512, C_N);
  transpose_w_kernel<<<CDIV(C_N*C_N,256),256,0,stream>>>(pi_w1, pi1T, C_N, C_N);

  // -------- stage 2: patch gather + local MLP (WMMA) --------
  patch_build_kernel<<<NTOK,256,0,stream>>>(x, soft, lin, gamma, patches);
  launch_gemm(stream, patches, C_N, C_N, nullptr,1,0, lr1T, C_N, lr_b1, nullptr,
              nullptr, hidden, C_N, NPAT, C_N, C_N, /*relu*/1);
  launch_gemm(stream, hidden, C_N, C_N, nullptr,1,0, lr2T, C_N, lr_b2, patches,
              nullptr, localb, C_N, NPAT, C_N, C_N, 0);

  // -------- stage 3: pooling + channel attention + fusion --------
  pool_kernel<<<NTOK,256,0,stream>>>(localb, meanb, maxb, sump);
  launch_gemm(stream, meanb, C_N, C_N, nullptr,1,0, ca1T, C_N, ca_b1, nullptr,
              nullptr, cah, 64, NTOK, C_N, 64, 1);
  launch_gemm(stream, cah, 64, 64, nullptr,1,0, ca2T, 64, ca_b2, nullptr,
              aw, nullptr, C_N, NTOK, 64, C_N, /*sigmoid*/2);
  weighted_kernel<<<CDIV((long)tokE,256),256,0,stream>>>(aw, sump, wgtb, (long)tokE);
  launch_gemm(stream, maxb, C_N, C_N, wgtb, 1, C_N, fusT, 512, fus_b, nullptr,
              token, nullptr, C_N, NTOK, 512, C_N, 0);
  posadd_kernel<<<CDIV((long)tokE,256),256,0,stream>>>(token, pos, posidx, tbf, (long)tokE);

  // -------- stage 4: transformer block (batched WMMA attention) --------
  launch_gemm(stream, tbf, C_N, C_N, nullptr,1,0, wqT, C_N, bq, nullptr, nullptr, qb,   C_N, NTOK, C_N, C_N, 0);
  launch_gemm(stream, tbf, C_N, C_N, nullptr,1,0, wkT, C_N, bk, nullptr, nullptr, kbuf, C_N, NTOK, C_N, C_N, 0);
  launch_gemm(stream, tbf, C_N, C_N, nullptr,1,0, wvT, C_N, bv, nullptr, nullptr, vb,   C_N, NTOK, C_N, C_N, 0);
  vT_kernel<<<CDIV((long)B_N*HEADS_N*DH_N*K_N,256),256,0,stream>>>(vb, vT);
  // scores[b,h] = q @ k^T   (32 batched GEMMs via gridDim.z)
  launch_gemm(stream, qb, DH_N, C_N, nullptr,1,0, kbuf, C_N, nullptr, nullptr,
              scores, nullptr, K_N, K_N, DH_N, K_N, 0,
              B_N*HEADS_N, HEADS_N,
              (long)K_N*C_N, DH_N, (long)K_N*C_N, DH_N,
              (long)HEADS_N*K_N*K_N, (long)K_N*K_N);
  attn_softmax_kernel<<<(unsigned)((long)B_N*HEADS_N*K_N),256,0,stream>>>(scores, att);
  // o[b,h] = att @ v
  launch_gemm(stream, att, K_N, K_N, nullptr,1,0, vT, K_N, nullptr, nullptr,
              obuf, nullptr, C_N, K_N, K_N, DH_N, 0,
              B_N*HEADS_N, HEADS_N,
              (long)HEADS_N*K_N*K_N, (long)K_N*K_N,
              (long)HEADS_N*DH_N*K_N, (long)DH_N*K_N,
              (long)K_N*C_N, DH_N);
  f2b_kernel<<<CDIV((long)tokE,256),256,0,stream>>>(obuf, obf, (long)tokE);
  launch_gemm(stream, obf, C_N, C_N, nullptr,1,0, woT, C_N, bo, nullptr,
              oproj, nullptr, C_N, NTOK, C_N, C_N, 0);
  ln_kernel<<<NTOK,256,0,stream>>>(token, oproj, ln1_g, ln1_b, t1, t1b);
  launch_gemm(stream, t1b, C_N, C_N, nullptr,1,0, ff1T, C_N, ff_b1, nullptr,
              nullptr, ffh, DFF_N, NTOK, C_N, DFF_N, 1);
  launch_gemm(stream, ffh, DFF_N, DFF_N, nullptr,1,0, ff2T, DFF_N, ff_b2, nullptr,
              ff2o, nullptr, C_N, NTOK, DFF_N, C_N, 0);
  ln_kernel<<<NTOK,256,0,stream>>>(t1, ff2o, ln2_g, ln2_b, t2, t2b);

  // -------- stage 5: gate + importance + scatter --------
  launch_gemm(stream, localb, C_N, C_N, t2b, P_N, C_N, gT, 512, g_b, nullptr,
              nullptr, gbuf, C_N, NPAT, 512, C_N, /*sigmoid*/2);
  fused_kernel<<<CDIV((long)patE,256),256,0,stream>>>(gbuf, t2b, localb, (long)patE);
  launch_gemm(stream, gbuf, C_N, C_N, nullptr,1,0, pi1T, C_N, pi_b1, nullptr,
              nullptr, hidden /*reuse*/, C_N, NPAT, C_N, C_N, 1);
  pi2_kernel<<<NPAT,256,0,stream>>>(hidden, pi_w2, pi_b2, imp);

  zero_kernel<<<CDIV((long)B_N*HW_N*C_N,256),256,0,stream>>>(fs, (long)B_N*HW_N*C_N);
  zero_kernel<<<CDIV((long)B_N*HW_N,256),256,0,stream>>>(wsum, (long)B_N*HW_N);
  scatter_kernel<<<NPAT,256,0,stream>>>(gbuf, imp, lin, fs, wsum);
  final_kernel<<<B_N*HW_N,256,0,stream>>>(fs, wsum, x, norm_g, norm_b, (float*)d_out);
}
